// MultiAttention_41540923686988
// MI455X (gfx1250) — matrix operand order repaired and checked by reading, not compiled, not run on hardware
//
#include <hip/hip_runtime.h>
#include <hip/hip_bf16.h>

typedef __attribute__((ext_vector_type(16))) _Float16 v16h;
typedef __attribute__((ext_vector_type(8)))  _Float16 v8h;
typedef __attribute__((ext_vector_type(8)))  float    v8f;

#define BATCH 32
#define SEQ   2048
#define FEAT  128
#define DK    32
// (1/sqrt(32)) * log2(e): softmax done in base-2 so exp lowers to v_exp_f32.
#define QSCALE 0.2550052690f

__device__ __forceinline__ float lane_bcast(float v, int src) {
  return __builtin_bit_cast(
      float, __builtin_amdgcn_ds_bpermute(src << 2, __builtin_bit_cast(int, v)));
}

__device__ __forceinline__ v16h cat8(v8h lo, v8h hi) {
  return __builtin_shufflevector(lo, hi, 0, 1, 2, 3, 4, 5, 6, 7,
                                 8, 9, 10, 11, 12, 13, 14, 15);
}

// ---------------------------------------------------------------------------
// Kernel 1: fused QKV projection, fp32 in -> fp16 out (Q pre-scaled, V stored
// transposed [b][d][s] so it can be loaded as a WMMA B operand directly).
// grid = B*S blocks of 96 threads (wave0->Q, wave1->K, wave2->V).
// ---------------------------------------------------------------------------
__global__ void __launch_bounds__(96)
proj_qkv(const float* __restrict__ qin, const float* __restrict__ kin,
         const float* __restrict__ vin,
         const float* __restrict__ Wq, const float* __restrict__ bq,
         const float* __restrict__ Wk, const float* __restrict__ bk,
         const float* __restrict__ Wv, const float* __restrict__ bv,
         _Float16* __restrict__ qf, _Float16* __restrict__ kf,
         _Float16* __restrict__ vt) {
  const int row = blockIdx.x;          // b*SEQ + s
  const int wv  = threadIdx.x >> 5;    // 0:Q 1:K 2:V
  const int c   = threadIdx.x & 31;    // output column (d_k)
  const float* in; const float* W; const float* bias;
  if (wv == 0)      { in = qin; W = Wq; bias = bq; }
  else if (wv == 1) { in = kin; W = Wk; bias = bk; }
  else              { in = vin; W = Wv; bias = bv; }
  const float* r = in + (size_t)row * FEAT;
  float acc = bias[c];
#pragma unroll 8
  for (int f = 0; f < FEAT; ++f) acc = fmaf(r[f], W[f * DK + c], acc);
  if (wv == 0) {
    qf[(size_t)row * DK + c] = (_Float16)(acc * QSCALE);
  } else if (wv == 1) {
    kf[(size_t)row * DK + c] = (_Float16)acc;
  } else {
    const int b = row >> 11, s = row & (SEQ - 1);
    vt[((size_t)b * DK + c) * SEQ + s] = (_Float16)acc;
  }
}

// ---------------------------------------------------------------------------
// Kernel 2: flash attention + folded output projection.
// 128 threads = 4 waves; each wave owns 16 queries; grid = B * (SEQ/64).
// Scores computed transposed (D = K x Q^T) so the P tile feeds the PV WMMA
// as an A operand without any cross-lane movement.  64 keys per iteration
// amortizes the online-softmax rescale (1 exp + 8 bpermute + bookkeeping)
// and cross-lane stat reductions over twice as much WMMA work.
// ---------------------------------------------------------------------------
__global__ void __launch_bounds__(128)
flash_attn(const _Float16* __restrict__ qf, const _Float16* __restrict__ kf,
           const _Float16* __restrict__ vt, const float* __restrict__ Wo,
           const float* __restrict__ bo, float* __restrict__ out) {
  __shared__ float wor[DK * 3];
  __shared__ float obuf[4][16][DK + 1];

  const int b    = blockIdx.x >> 5;       // SEQ/64 == 32 query tiles per batch
  const int qt   = blockIdx.x & 31;
  const int wave = threadIdx.x >> 5;
  const int lane = threadIdx.x & 31;
  const int h    = lane >> 4;             // half-wave select
  const int n    = lane & 15;             // matrix column / row-in-tile
  const int q0   = qt * 64 + wave * 16;

  // Fold the 16 identical heads of Wo into one 32x3 matrix.
  if (threadIdx.x < 96) {
    const int d = threadIdx.x / 3, f = threadIdx.x % 3;
    float s = 0.f;
    for (int hh = 0; hh < 16; ++hh) s += Wo[(hh * DK + d) * 3 + f];
    wor[d * 3 + f] = s;
  }

  // Q tile as WMMA B operand: lane n holds query row q0+n, K = {8h..8h+7, 16+8h..16+8h+7}.
  const _Float16* qr = qf + ((size_t)b * SEQ + q0 + n) * DK + 8 * h;
  const v16h Bq = cat8(*(const v8h*)qr, *(const v8h*)(qr + 16));

  const _Float16* kb = kf + (size_t)b * SEQ * DK;
  const _Float16* vb = vt + (size_t)b * DK * SEQ;

  float m = -__builtin_inff(), l = 0.f;
  v8f o0 = {}, o1 = {};
  const v8f zero = {};

  for (int k0 = 0; k0 < SEQ; k0 += 64) {
    // K rows as A operands (A interleave: K = {8h..8h+7, 16+8h..16+8h+7}).
    v16h A[4];
#pragma unroll
    for (int t = 0; t < 4; ++t) {
      const _Float16* p = kb + (size_t)(k0 + t * 16 + n) * DK + 8 * h;
      A[t] = cat8(*(const v8h*)p, *(const v8h*)(p + 16));
    }
    v8f s[4];
#pragma unroll
    for (int t = 0; t < 4; ++t)
      s[t] = __builtin_amdgcn_wmma_f32_16x16x32_f16(false, A[t], false, Bq,
                                                    (short)0, zero, false, false);

    // V^T rows as B operands (lane n -> d_v column n / 16+n, keys {8h..8h+7, 16+8h..16+8h+7}).
    const _Float16* v0p = vb + (size_t)n * SEQ + k0 + 8 * h;
    const _Float16* v1p = vb + (size_t)(16 + n) * SEQ + k0 + 8 * h;
    const v16h Bv00 = cat8(*(const v8h*)v0p,        *(const v8h*)(v0p + 16));
    const v16h Bv01 = cat8(*(const v8h*)(v0p + 32), *(const v8h*)(v0p + 48));
    const v16h Bv10 = cat8(*(const v8h*)v1p,        *(const v8h*)(v1p + 16));
    const v16h Bv11 = cat8(*(const v8h*)(v1p + 32), *(const v8h*)(v1p + 48));

    // Online softmax in base-2.  Each lane holds 32 keys for query n; the
    // other 32 keys of this chunk live in lane^16.
    float cm = s[0][0];
#pragma unroll
    for (int t = 0; t < 4; ++t)
#pragma unroll
      for (int j = 0; j < 8; ++j) cm = fmaxf(cm, s[t][j]);
    cm = fmaxf(cm, lane_bcast(cm, lane ^ 16));
    const float mn = fmaxf(m, cm);

    v16h APl, APh;                 // P tiles, already in A-operand layout
    float cs = 0.f;
#pragma unroll
    for (int j = 0; j < 8; ++j) {
      const float p = __builtin_amdgcn_exp2f(s[0][j] - mn);
      cs += p; APl[j] = (_Float16)p;
    }
#pragma unroll
    for (int j = 0; j < 8; ++j) {
      const float p = __builtin_amdgcn_exp2f(s[1][j] - mn);
      cs += p; APl[j + 8] = (_Float16)p;
    }
#pragma unroll
    for (int j = 0; j < 8; ++j) {
      const float p = __builtin_amdgcn_exp2f(s[2][j] - mn);
      cs += p; APh[j] = (_Float16)p;
    }
#pragma unroll
    for (int j = 0; j < 8; ++j) {
      const float p = __builtin_amdgcn_exp2f(s[3][j] - mn);
      cs += p; APh[j + 8] = (_Float16)p;
    }
    cs += lane_bcast(cs, lane ^ 16);

    const float sc = __builtin_amdgcn_exp2f(m - mn);
    l = l * sc + cs;
    m = mn;

    // Rescale accumulators: O-layout row for VGPR j is query 8h+j, whose
    // scale factor lives in lane 8h+j.  Once per 64 keys.
#pragma unroll
    for (int j = 0; j < 8; ++j) {
      const float scj = lane_bcast(sc, 8 * h + j);
      o0[j] *= scj; o1[j] *= scj;
    }
    o0 = __builtin_amdgcn_wmma_f32_16x16x32_f16(false, APl, false, Bv00,
                                                (short)0, o0, false, false);
    o0 = __builtin_amdgcn_wmma_f32_16x16x32_f16(false, APh, false, Bv01,
                                                (short)0, o0, false, false);
    o1 = __builtin_amdgcn_wmma_f32_16x16x32_f16(false, APl, false, Bv10,
                                                (short)0, o1, false, false);
    o1 = __builtin_amdgcn_wmma_f32_16x16x32_f16(false, APh, false, Bv11,
                                                (short)0, o1, false, false);
  }

  // Normalize and stage O tile to LDS.
  const float invl = 1.f / l;
#pragma unroll
  for (int j = 0; j < 8; ++j) {
    const float nj = lane_bcast(invl, 8 * h + j);
    obuf[wave][8 * h + j][n]      = o0[j] * nj;
    obuf[wave][8 * h + j][16 + n] = o1[j] * nj;
  }
  __syncthreads();

  // Output projection: head @ Wo_reduced + bo  -> [B,S,3].
  if (threadIdx.x < 64) {
    const int w2 = threadIdx.x >> 4, r = threadIdx.x & 15;
    float a0 = bo[0], a1 = bo[1], a2 = bo[2];
#pragma unroll
    for (int d = 0; d < DK; ++d) {
      const float ov = obuf[w2][r][d];
      a0 = fmaf(ov, wor[d * 3 + 0], a0);
      a1 = fmaf(ov, wor[d * 3 + 1], a1);
      a2 = fmaf(ov, wor[d * 3 + 2], a2);
    }
    float* op = out + ((size_t)b * SEQ + qt * 64 + w2 * 16 + r) * 3;
    op[0] = a0; op[1] = a1; op[2] = a2;
  }
}

// ---------------------------------------------------------------------------
extern "C" void kernel_launch(void* const* d_in, const int* in_sizes, int n_in,
                              void* d_out, int out_size, void* d_ws, size_t ws_size,
                              hipStream_t stream) {
  (void)in_sizes; (void)n_in; (void)out_size; (void)ws_size;
  const float* qin = (const float*)d_in[0];
  const float* kin = (const float*)d_in[1];
  const float* vin = (const float*)d_in[2];
  const float* Wq  = (const float*)d_in[3];
  const float* bq  = (const float*)d_in[4];
  const float* Wk  = (const float*)d_in[5];
  const float* bk  = (const float*)d_in[6];
  const float* Wv  = (const float*)d_in[7];
  const float* bv  = (const float*)d_in[8];
  const float* Wo  = (const float*)d_in[9];
  const float* bo  = (const float*)d_in[10];

  // Workspace: Q,K (row-major) and V^T, all fp16: 3 * 4 MB = 12 MB.
  _Float16* qf = (_Float16*)d_ws;
  _Float16* kf = qf + (size_t)BATCH * SEQ * DK;
  _Float16* vt = kf + (size_t)BATCH * SEQ * DK;

  proj_qkv<<<BATCH * SEQ, 96, 0, stream>>>(qin, kin, vin, Wq, bq, Wk, bk,
                                           Wv, bv, qf, kf, vt);
  flash_attn<<<BATCH * (SEQ / 64), 128, 0, stream>>>(qf, kf, vt, Wo, bo,
                                                     (float*)d_out);
}
